// QuantizedLinear_44607530336759
// MI455X (gfx1250) — matrix operand order repaired and checked by reading, not compiled, not run on hardware
//
#include <hip/hip_runtime.h>
#include <stdint.h>

typedef _Float16 h16;
typedef __attribute__((ext_vector_type(16))) _Float16 v16h;
typedef __attribute__((ext_vector_type(8)))  _Float16 v8h;
typedef __attribute__((ext_vector_type(8)))  float    v8f;

#define BM 128
#define BN 256
#define BK 32
#define BKP 40   // padded LDS row stride (halves): 80B, 16B-aligned, 64-bank conflict-free

union Frag { v16h v; v8h h[2]; };

// out[m][n] = scale[n] * sum_k A[m][k] * (fp16)Wq[n][k]
__global__ __launch_bounds__(256)
void qlinear_wmma_kernel(const h16* __restrict__ A,     // [M][K] fp16
                         const int* __restrict__ Wq,    // [N][K] int32 (int8 values)
                         const h16* __restrict__ scale, // [N] fp16
                         float* __restrict__ C,         // [M][N] f32
                         int M, int Nn, int Kk)
{
    __shared__ __align__(16) h16 As[2][BM][BKP];   // 20 KB
    __shared__ __align__(16) h16 Bs[2][BN][BKP];   // 40 KB

    const int tid   = threadIdx.x;
    const int lane  = tid & 31;
    const int wave  = tid >> 5;
    const int wm    = wave & 1;    // 2 waves along M
    const int wn    = wave >> 1;   // 4 waves along N
    const int lrow  = lane & 15;
    const int lhalf = lane >> 4;

    const int m0 = blockIdx.y * BM;
    const int n0 = blockIdx.x * BN;

    // ---- A staging: async global->LDS, 512 x 16B chunks per tile, 2/thread ----
    const int c0row = tid >> 2;            // chunk c: row = c/4, col = (c%4)*8 halves
    const int c0col = (tid & 3) * 8;
    const int c1row = c0row + 64;          // chunk c+256
    const size_t aBase0 = (size_t)(m0 + c0row) * Kk + c0col;
    const size_t aBase1 = (size_t)(m0 + c1row) * Kk + c0col;

    // ---- W staging: thread t owns one full 32-int row of the 256-row tile ----
    const size_t wBase = (size_t)(n0 + tid) * Kk;

    v8f acc[4][4] = {};
    int4 wReg[8];

    auto async_load_A = [&](int kt, int buf) {
        // low 32 bits of a generic LDS pointer == LDS byte offset (VDST operand)
        uint32_t l0 = (uint32_t)(uintptr_t)&As[buf][c0row][c0col];
        uint32_t l1 = (uint32_t)(uintptr_t)&As[buf][c1row][c0col];
        uint64_t g0 = (uint64_t)(uintptr_t)(A + aBase0 + (size_t)kt * BK);
        uint64_t g1 = (uint64_t)(uintptr_t)(A + aBase1 + (size_t)kt * BK);
        asm volatile("global_load_async_to_lds_b128 %0, %1, off"
                     :: "v"(l0), "v"(g0) : "memory");
        asm volatile("global_load_async_to_lds_b128 %0, %1, off"
                     :: "v"(l1), "v"(g1) : "memory");
    };
    auto wait_async = [&]() {
        asm volatile("s_wait_asynccnt 0x0" ::: "memory");
    };

    auto load_W_regs = [&](int kt) {
        const int4* wp = (const int4*)(Wq + wBase + (size_t)kt * BK);
        #pragma unroll
        for (int j = 0; j < 8; ++j) wReg[j] = wp[j];
    };

    auto store_W_lds = [&](int buf) {
        h16 wh[32];
        const int* wi = (const int*)&wReg[0];
        #pragma unroll
        for (int j = 0; j < 32; ++j) wh[j] = (h16)(float)wi[j];  // exact for int8 range
        #pragma unroll
        for (int j = 0; j < 4; ++j)
            *(v8h*)&Bs[buf][tid][j * 8] = *(const v8h*)&wh[j * 8];
    };

    const int nkt = Kk / BK;   // 128 K-steps

    // prologue: tile 0
    async_load_A(0, 0);
    load_W_regs(0);
    store_W_lds(0);
    wait_async();

    for (int kt = 0; kt < nkt; ++kt) {
        const int buf = kt & 1;
        __syncthreads();                       // As[buf]/Bs[buf] visible to all waves
        if (kt + 1 < nkt) {
            async_load_A(kt + 1, buf ^ 1);     // ASYNCcnt path, no VGPR staging
            load_W_regs(kt + 1);               // LOADcnt path
        }

        // A fragments (16x32 fp16, ISA layout): lanes 0-15: K 0-7 & 16-23; lanes 16-31: K 8-15 & 24-31
        Frag afrag[4];
        #pragma unroll
        for (int mi = 0; mi < 4; ++mi) {
            const int r  = wm * 64 + mi * 16 + lrow;
            const int kA = lhalf * 8;
            afrag[mi].h[0] = *(const v8h*)&As[buf][r][kA];
            afrag[mi].h[1] = *(const v8h*)&As[buf][r][kA + 16];
        }
        // B fragments (32x16 fp16): lane = output column; lanes 0-15: K 0-7 & 16-23; lanes 16-31: K 8-15 & 24-31
        Frag bfrag[4];
        #pragma unroll
        for (int ni = 0; ni < 4; ++ni) {
            const int c  = wn * 64 + ni * 16 + lrow;
            const int kB = lhalf * 8;
            bfrag[ni].h[0] = *(const v8h*)&Bs[buf][c][kB];
            bfrag[ni].h[1] = *(const v8h*)&Bs[buf][c][kB + 16];
        }

        #pragma unroll
        for (int mi = 0; mi < 4; ++mi)
            #pragma unroll
            for (int ni = 0; ni < 4; ++ni)
                acc[mi][ni] = __builtin_amdgcn_wmma_f32_16x16x32_f16(
                    false, afrag[mi].v, false, bfrag[ni].v,
                    (short)0, acc[mi][ni], false, false);

        if (kt + 1 < nkt) {
            store_W_lds(buf ^ 1);              // dequantized weights -> other buffer
            wait_async();                      // our A chunks landed before next barrier
        }
    }

    // Epilogue: per-channel scale, f32 store.
    // C/D layout: lanes 0-15 -> N=lane, M = vgpr idx; lanes 16-31 -> M+8.
    #pragma unroll
    for (int ni = 0; ni < 4; ++ni) {
        const int nG = n0 + wn * 64 + ni * 16 + lrow;
        const float s = (float)scale[nG];
        #pragma unroll
        for (int mi = 0; mi < 4; ++mi) {
            const int mBase = m0 + wm * 64 + mi * 16 + lhalf * 8;
            #pragma unroll
            for (int r = 0; r < 8; ++r)
                C[(size_t)(mBase + r) * Nn + nG] = acc[mi][ni][r] * s;
        }
    }
}

extern "C" void kernel_launch(void* const* d_in, const int* in_sizes, int n_in,
                              void* d_out, int out_size, void* d_ws, size_t ws_size,
                              hipStream_t stream) {
    const h16* A   = (const h16*)d_in[0];   // input  fp16 [B,S,K] -> [M,K]
    const int* Wq  = (const int*)d_in[1];   // weight int32 [N,K]
    const h16* sc  = (const h16*)d_in[2];   // scale  fp16 [N]
    float*     out = (float*)d_out;         // [M,N]

    const int Nn = in_sizes[2];             // 11008
    const int Kk = in_sizes[1] / Nn;        // 4096
    const int M  = in_sizes[0] / Kk;        // 8192

    dim3 grid(Nn / BN, M / BM);             // 43 x 64 workgroups
    qlinear_wmma_kernel<<<grid, 256, 0, stream>>>(A, Wq, sc, out, M, Nn, Kk);
}
